// Model_19095424598430
// MI455X (gfx1250) — compile-verified, matrix operand order checked by reading, not run on hardware
//
#include <hip/hip_runtime.h>
#include <math.h>

// ---------------------------------------------------------------------------
// Sequential RNN (h = tanh(x_t*U + h@W + b)) on MI455X / gfx1250.
//   * 16 workgroups x 16 batch rows (rows independent -> no inter-WG sync).
//   * Weight-stationary bf16 W: K-fragments 3..15 in VGPRs (416 regs/wave),
//     0..2 in LDS (96KB), packed once into WMMA B-fragment lane layout.
//   * Per step: 64x v_wmma_f32_16x16x32_bf16 per wave, A fragments software-
//     pipelined from double-buffered LDS h, ONE barrier per step.
//   * Hardware v_tanh_f32 when available (__has_builtin guarded).
// ---------------------------------------------------------------------------

typedef __attribute__((ext_vector_type(16))) __bf16 v16bf;
typedef __attribute__((ext_vector_type(8)))  float  v8f;

#define Bsz 256
#define Tn  784
#define Hn  512
#define Cn  10

#define ROWS 16                 // batch rows per workgroup (M tile)
#define NWAVE 8                 // waves per workgroup (wave32)
#define NTHREADS (NWAVE * 32)
#define TILES_PER_WAVE 4        // 4 x 16 = 64 output columns per wave
#define KP 520                  // padded h row stride in halves (bank spread)
#define LDSF 3                  // K-fragments (32 wide) of W kept in LDS
#define REGF 13                 // K-fragments of W kept in VGPRs

#if __has_builtin(__builtin_amdgcn_tanhf)
#define fast_tanh(x) __builtin_amdgcn_tanhf(x)
#elif __has_builtin(__builtin_amdgcn_tanh_f32)
#define fast_tanh(x) __builtin_amdgcn_tanh_f32(x)
#else
#define fast_tanh(x) tanhf(x)
#endif

__device__ __forceinline__ unsigned short f32_to_bf16(float f) {
  unsigned int u = __float_as_uint(f);
  u += 0x7FFFu + ((u >> 16) & 1u);          // round-to-nearest-even
  return (unsigned short)(u >> 16);
}
__device__ __forceinline__ float bf16_to_f32(unsigned short h) {
  return __uint_as_float(((unsigned int)h) << 16);
}

// 16-bit A/B fragment element -> local K index (ISA 7.12.2 layout):
// VGPR j holds K pairs: j=0..3 -> K = grp*8 + 2j,2j+1 ; j=4..7 -> +16.
__device__ __forceinline__ int kloc(int e, int grp) {
  int j = e >> 1;
  return ((j & 4) ? 16 : 0) + grp * 8 + (j & 3) * 2 + (e & 1);
}

union FragU {
  uint4 q[2];
  v16bf v;
};

// ---------------------------------------------------------------------------
// Pack W (f32, [K=512][N=512] row-major) into bf16 WMMA B-fragment order:
//   halves index = (((nt*16 + f)*32 + lane)*16 + e)
//   nt = 16-column tile, f = K-block of 32, lane -> N = nt*16 + (lane&15).
// ---------------------------------------------------------------------------
__global__ void pack_w_kernel(const float* __restrict__ W,
                              unsigned short* __restrict__ wp) {
  int idx = blockIdx.x * blockDim.x + threadIdx.x;   // (nt*16 + f)*32 + lane
  if (idx >= 32 * 16 * 32) return;
  int lane = idx & 31;
  int f    = (idx >> 5) & 15;
  int nt   = idx >> 9;
  int n    = nt * 16 + (lane & 15);
  int grp  = lane >> 4;
  unsigned short* dst = wp + idx * 16;
#pragma unroll
  for (int e = 0; e < 16; ++e) {
    int k = f * 32 + kloc(e, grp);
    dst[e] = f32_to_bf16(W[k * Hn + n]);
  }
}

// ---------------------------------------------------------------------------
// Main recurrent kernel: one workgroup = 16 batch rows, all 512 hidden cols.
// ---------------------------------------------------------------------------
__global__ void __launch_bounds__(NTHREADS, 1) rnn_rowwise_kernel(
    const float* __restrict__ inputs,          // [B,T]
    const int*   __restrict__ yv,              // [B]
    const int*   __restrict__ order,           // [T]
    const float* __restrict__ Uvec,            // [H]
    const float* __restrict__ bvec,            // [H]
    const float* __restrict__ lin_w,           // [H,C]
    const float* __restrict__ lin_b,           // [C]
    const unsigned short* __restrict__ wp,     // packed W bf16 fragments
    float* __restrict__ out)                   // out[0]=loss, out[1]=correct
{
  __shared__ __align__(32) unsigned short wLds[32 * LDSF * 32 * 16]; // 96 KB
  __shared__ __align__(32) unsigned short hbuf[2 * ROWS * KP];       // 2x16.6KB

  const int tid  = threadIdx.x;
  const int lane = tid & 31;
  const int wave = tid >> 5;
  const int grp  = lane >> 4;
  const int nl   = lane & 15;
  const int gm0  = blockIdx.x * ROWS;

  // ---- stage LDS-resident W fragments (K blocks 0..LDSF-1, all tiles) ----
  for (int i = tid; i < 32 * LDSF * 32; i += NTHREADS) {
    int nt = i / (LDSF * 32);
    int r  = i - nt * (LDSF * 32);
    int f  = r >> 5;
    int ln = r & 31;
    const uint4* src = (const uint4*)(wp + ((nt * 16 + f) * 32 + ln) * 16);
    uint4* dst = (uint4*)(wLds + i * 16);
    dst[0] = src[0];
    dst[1] = src[1];
  }
  // ---- h0 = 0 (both buffers) ----
  for (int i = tid; i < 2 * ROWS * KP; i += NTHREADS) hbuf[i] = 0;

  // ---- register-resident W fragments (K blocks LDSF..15) ----
  v16bf wreg[TILES_PER_WAVE][REGF];
#pragma unroll
  for (int ti = 0; ti < TILES_PER_WAVE; ++ti) {
    int nt = wave * TILES_PER_WAVE + ti;
#pragma unroll
    for (int f = 0; f < REGF; ++f) {
      FragU fu;
      const uint4* src =
          (const uint4*)(wp + ((nt * 16 + (f + LDSF)) * 32 + lane) * 16);
      fu.q[0] = src[0];
      fu.q[1] = src[1];
      wreg[ti][f] = fu.v;
    }
  }

  // per-lane U / bias for the owned column tiles (n = tile*16 + nl)
  float un[TILES_PER_WAVE], bn[TILES_PER_WAVE];
#pragma unroll
  for (int ti = 0; ti < TILES_PER_WAVE; ++ti) {
    int n = (wave * TILES_PER_WAVE + ti) * 16 + nl;
    un[ti] = Uvec[n];
    bn[ti] = bvec[n];
  }

  __syncthreads();

  const int m_base = grp * 8;   // C-matrix rows this lane holds: m_base..+7
  const int m      = nl;        // A-matrix row this lane supplies

#pragma unroll 1
  for (int t = 0; t < Tn; ++t) {
    const unsigned short* hRd = hbuf + (t & 1) * (ROWS * KP);
    unsigned short*       hWr = hbuf + ((t + 1) & 1) * (ROWS * KP);

    // hoist the per-step global traffic above the WMMA chain
    const int col = order[t];
    float xv[8];
#pragma unroll
    for (int r = 0; r < 8; ++r)
      xv[r] = inputs[(gm0 + m_base + r) * Tn + col];

    v8f acc[TILES_PER_WAVE];
#pragma unroll
    for (int ti = 0; ti < TILES_PER_WAVE; ++ti) {
      v8f z = {};
      acc[ti] = z;
    }

    // software-pipelined A fragments (prefetch f+1 under WMMA f)
    FragU aCur, aNxt;
    aCur.q[0] = *(const uint4*)(hRd + m * KP + grp * 8);
    aCur.q[1] = *(const uint4*)(hRd + m * KP + 16 + grp * 8);
#pragma unroll
    for (int f = 0; f < 16; ++f) {
      if (f < 15) {
        aNxt.q[0] = *(const uint4*)(hRd + m * KP + (f + 1) * 32 + grp * 8);
        aNxt.q[1] = *(const uint4*)(hRd + m * KP + (f + 1) * 32 + 16 + grp * 8);
      }
#pragma unroll
      for (int ti = 0; ti < TILES_PER_WAVE; ++ti) {
        v16bf bf;
        if (f < LDSF) {
          int nt = wave * TILES_PER_WAVE + ti;
          bf = *(const v16bf*)(wLds + ((nt * LDSF + f) * 32 + lane) * 16);
        } else {
          bf = wreg[ti][f - LDSF];
        }
        acc[ti] = __builtin_amdgcn_wmma_f32_16x16x32_bf16(
            false, aCur.v, false, bf, (short)0, acc[ti], false, false);
      }
      if (f < 15) aCur = aNxt;
    }

    // epilogue: pre-activation + hardware tanh, write new h to other buffer
#pragma unroll
    for (int ti = 0; ti < TILES_PER_WAVE; ++ti) {
      int n = (wave * TILES_PER_WAVE + ti) * 16 + nl;
#pragma unroll
      for (int r = 0; r < 8; ++r) {
        float v = acc[ti][r] + xv[r] * un[ti] + bn[ti];
        hWr[(m_base + r) * KP + n] = f32_to_bf16(fast_tanh(v));
      }
    }
    __syncthreads();   // one barrier per step (double-buffered h)
  }

  // final h lives in buffer 0 (T even); buffer 1 is free -> logits scratch
  const unsigned short* hFin = hbuf;
  float* logitsS = (float*)(hbuf + ROWS * KP);

  // ---- classifier head: logits[m][c] = h[m] . lin_w[:,c] + lin_b[c] ----
  if (tid < ROWS * Cn) {
    int mm = tid & 15;
    int cc = tid >> 4;          // < 10 since tid < 160
    float s = lin_b[cc];
    for (int k = 0; k < Hn; ++k)
      s += bf16_to_f32(hFin[mm * KP + k]) * lin_w[k * Cn + cc];
    logitsS[mm * Cn + cc] = s;
  }
  __syncthreads();

  // ---- per-row log-softmax, loss and argmax; atomic partials ----
  if (tid < ROWS) {
    int label = yv[gm0 + tid];
    float mx = logitsS[tid * Cn];
    int am = 0;
    for (int c = 1; c < Cn; ++c) {
      float lv = logitsS[tid * Cn + c];
      if (lv > mx) { mx = lv; am = c; }
    }
    float se = 0.0f;
    for (int c = 0; c < Cn; ++c) se += __expf(logitsS[tid * Cn + c] - mx);
    float lse = mx + __logf(se);
    float lossm = lse - logitsS[tid * Cn + label];
    atomicAdd(&out[0], lossm * (1.0f / (float)Bsz));
    atomicAdd(&out[1], (am == label) ? 1.0f : 0.0f);
  }
}

// ---------------------------------------------------------------------------
extern "C" void kernel_launch(void* const* d_in, const int* in_sizes, int n_in,
                              void* d_out, int out_size, void* d_ws,
                              size_t ws_size, hipStream_t stream) {
  const float* inputs = (const float*)d_in[0];
  const int*   yv     = (const int*)d_in[1];
  const int*   order  = (const int*)d_in[2];
  const float* Uvec   = (const float*)d_in[3];
  const float* Wm     = (const float*)d_in[4];
  const float* bvec   = (const float*)d_in[5];
  const float* lin_w  = (const float*)d_in[6];
  const float* lin_b  = (const float*)d_in[7];
  float* out = (float*)d_out;
  unsigned short* wp = (unsigned short*)d_ws;   // 512 KB packed bf16 W

  (void)in_sizes; (void)n_in; (void)ws_size;

  hipMemsetAsync(d_out, 0, (size_t)out_size * sizeof(float), stream);
  pack_w_kernel<<<(32 * 16 * 32 + 255) / 256, 256, 0, stream>>>(Wm, wp);
  rnn_rowwise_kernel<<<Bsz / ROWS, NTHREADS, 0, stream>>>(
      inputs, yv, order, Uvec, bvec, lin_w, lin_b, wp, out);
}
